// DAWN_45140106281448
// MI455X (gfx1250) — matrix operand order repaired and checked by reading, not compiled, not run on hardware
//
#include <hip/hip_runtime.h>
#include <cstdint>

// ---------------- problem dims ----------------
#define T_TOK   2048          // B*S
#define DMODEL  1024
#define RANK    256
#define NNEUR   64
#define NHEAD   8
#define DHEAD   32
#define SEQ     512
#define NKNOW   8192
#define CAP     2048          // max tokens per neuron bin

typedef __bf16 v16bf __attribute__((ext_vector_type(16)));
typedef __bf16 bf8   __attribute__((ext_vector_type(8)));
typedef float  v8f   __attribute__((ext_vector_type(8)));

// ---------------- utility kernels ----------------
__global__ void zero_f32_kernel(float* __restrict__ p, int n) {
    int i = blockIdx.x * 256 + threadIdx.x;
    if (i < n) p[i] = 0.f;
}

__global__ void copy_f32_kernel(float* __restrict__ d, const float* __restrict__ s, int n) {
    int i = blockIdx.x * 256 + threadIdx.x;
    if (i < n) d[i] = s[i];
}

__global__ void f32_to_bf16_kernel(__bf16* __restrict__ d, const float* __restrict__ s, int n) {
    int i = blockIdx.x * 256 + threadIdx.x;
    if (i < n) d[i] = (__bf16)s[i];
}

// ---------------- weight prepack: fp32 -> bf16 in WMMA per-lane fragment layout ----------
// Packed layout: P[((n*KT + kt)*SL + slice)*512 + lane*16 + e]
//   kt = k/32, slice = c/16, lane = (k%32 < 16) ? c%16 : 16 + c%16, e = k%16
// bT=0: source W[n][Kdim][Ndim] (c fastest for coalescing)
// bT=1: source W[c][Kdim]       (k fastest for coalescing, n==0)
__global__ void prepack_kernel(const float* __restrict__ W, __bf16* __restrict__ P,
                               int Kdim, int Ndim, int bT) {
    long long idx = (long long)blockIdx.x * 256 + threadIdx.x;
    int k, c; long long n; float v;
    if (bT) {
        k = (int)(idx % Kdim);
        long long rest = idx / Kdim;
        c = (int)(rest % Ndim);
        n = rest / Ndim;
        v = W[(n * Ndim + c) * (long long)Kdim + k];
    } else {
        c = (int)(idx % Ndim);
        long long rest = idx / Ndim;
        k = (int)(rest % Kdim);
        n = rest / Kdim;
        v = W[idx];
    }
    int kt = k >> 5, r2 = k & 31, slice = c >> 4, cl = c & 15;
    int lane2 = (r2 < 16) ? cl : (16 + cl);
    int e2 = r2 & 15;
    long long KT = Kdim >> 5, SL = Ndim >> 4;
    P[(((n * KT + kt) * SL + slice) << 9) + lane2 * 16 + e2] = (__bf16)v;
}

// ---------------- LayerNorm -> bf16 (D=1024, 256 thr/blk, 1 blk/token) ----------------
__global__ void ln_bf16_kernel(const float* __restrict__ X, const float* __restrict__ g,
                               const float* __restrict__ b, __bf16* __restrict__ Hbf) {
    __shared__ float red[256];
    int t = blockIdx.x, tid = threadIdx.x;
    const float* xr = X + (size_t)t * DMODEL;
    float vals[4], s = 0.f;
#pragma unroll
    for (int i = 0; i < 4; i++) { vals[i] = xr[tid + 256 * i]; s += vals[i]; }
    red[tid] = s; __syncthreads();
    for (int off = 128; off > 0; off >>= 1) { if (tid < off) red[tid] += red[tid + off]; __syncthreads(); }
    float mu = red[0] * (1.f / DMODEL);
    __syncthreads();
    float v = 0.f;
#pragma unroll
    for (int i = 0; i < 4; i++) { float d = vals[i] - mu; v += d * d; }
    red[tid] = v; __syncthreads();
    for (int off = 128; off > 0; off >>= 1) { if (tid < off) red[tid] += red[tid + off]; __syncthreads(); }
    float rs = rsqrtf(red[0] * (1.f / DMODEL) + 1e-5f);
#pragma unroll
    for (int i = 0; i < 4; i++) {
        int d = tid + 256 * i;
        Hbf[(size_t)t * DMODEL + d] = (__bf16)((vals[i] - mu) * rs * g[d] + b[d]);
    }
}

// ---------------- WMMA GEMM on prepacked bf16 B, barrier-free inner loop ----------------
// block = 128 threads (4 waves); block tile M=16, N=64 (one 16-col slice per wave).
// GATHER=false: A row = token m0+row, plain store Out = scale*acc
// GATHER=true : neuron = blockIdx.z, rows gathered from bins, atomic scatter w*acc
template <bool GATHER>
__global__ void wmma_gemm_kernel(const __bf16* __restrict__ A, const __bf16* __restrict__ Bp,
                                 float* __restrict__ Out,
                                 const int* __restrict__ btok, const float* __restrict__ bw,
                                 const int* __restrict__ bcnt,
                                 int Kdim, int Ndim, int ldout, float scale) {
    __shared__ int   sTok[16];
    __shared__ float sW[16];
    int tid = threadIdx.x, wave = tid >> 5, lane = tid & 31;
    int m0 = blockIdx.y * 16, ncol0 = blockIdx.x * 64;

    int cnt = T_TOK, neuron = 0;
    if (GATHER) {
        neuron = blockIdx.z;
        cnt = bcnt[neuron];
        if (m0 >= cnt) return;
        if (tid < 16) {
            int gi = m0 + tid;
            if (gi < cnt) {
                sTok[tid] = btok[(size_t)neuron * CAP + gi];
                sW[tid]   = bw  [(size_t)neuron * CAP + gi];
            } else { sTok[tid] = 0; sW[tid] = 0.f; }
        }
        __syncthreads();
    }

    int row = lane & 15, kbase = (lane < 16) ? 0 : 16;
    int tok = GATHER ? sTok[row] : (m0 + row);
    const __bf16* Arow = A + (size_t)tok * Kdim + kbase;

    int KT = Kdim >> 5, SL = Ndim >> 4;
    int slice = blockIdx.x * 4 + wave;
    const __bf16* Pb = Bp + (((size_t)neuron * KT * SL + slice) << 9) + lane * 16;
    size_t kstride = (size_t)SL << 9;

    v8f acc = {0.f, 0.f, 0.f, 0.f, 0.f, 0.f, 0.f, 0.f};
#pragma unroll 2
    for (int kt = 0; kt < KT; kt++) {
        const __bf16* ak = Arow + (kt << 5);
        bf8 a0 = *reinterpret_cast<const bf8*>(ak);
        bf8 a1 = *reinterpret_cast<const bf8*>(ak + 8);
        v16bf Bf = *reinterpret_cast<const v16bf*>(Pb + kt * kstride);
        if (kt + 2 < KT) __builtin_prefetch(Pb + (kt + 2) * kstride, 0, 3);
        v16bf Af;
#pragma unroll
        for (int i = 0; i < 8; i++) { Af[i] = a0[i]; Af[i + 8] = a1[i]; }
        acc = __builtin_amdgcn_wmma_f32_16x16x32_bf16(false, Af, false, Bf, (short)0, acc, false, false);
    }

    int ncl = lane & 15;
    int mb  = (lane < 16) ? 0 : 8;
    int nc  = ncol0 + wave * 16 + ncl;
#pragma unroll
    for (int g2 = 0; g2 < 8; g2++) {
        int m = mb + g2;
        if (GATHER) {
            if (m0 + m < cnt)
                atomicAdd(&Out[(size_t)sTok[m] * ldout + nc], sW[m] * acc[g2]);
        } else {
            Out[(size_t)(m0 + m) * ldout + nc] = scale * acc[g2];
        }
    }
}

// ---------------- top-k + softmax over 64 router scores ----------------
__global__ void topk_softmax_kernel(const float* __restrict__ scores, int topk,
                                    int* __restrict__ tidx, float* __restrict__ tw) {
    int t = blockIdx.x, r = blockIdx.y;
    if (threadIdx.x != 0) return;
    const float* sc = scores + ((size_t)r * T_TOK + t) * NNEUR;
    float bestv[8]; int besti[8];
    unsigned long long used = 0ull;
    for (int k = 0; k < topk; k++) {
        float bv = -1e30f; int bi = 0;
        for (int n = 0; n < NNEUR; n++) {
            if ((used >> n) & 1ull) continue;
            float x = sc[n];
            if (x > bv) { bv = x; bi = n; }
        }
        used |= 1ull << bi;
        bestv[k] = bv; besti[k] = bi;
    }
    float mx = bestv[0], ssum = 0.f, e[8];
    for (int k = 0; k < topk; k++) { e[k] = __expf(bestv[k] - mx); ssum += e[k]; }
    float inv = 1.f / ssum;
    for (int k = 0; k < topk; k++) {
        tidx[((size_t)r * T_TOK + t) * topk + k] = besti[k];
        tw  [((size_t)r * T_TOK + t) * topk + k] = e[k] * inv;
    }
}

// ---------------- deterministic per-neuron binning (ordered prefix scan) ----------------
__global__ void bin_kernel(const int* __restrict__ tidx, const float* __restrict__ tw,
                           int topk, int* __restrict__ btok, float* __restrict__ bw,
                           int* __restrict__ bcnt) {
    __shared__ int sc[256];
    int neuron = blockIdx.x, r = blockIdx.y, tid = threadIdx.x;
    const int*   ti  = tidx + (size_t)r * T_TOK * topk;
    const float* twr = tw   + (size_t)r * T_TOK * topk;
    int t0 = tid * 8, cnt = 0;
    for (int t = t0; t < t0 + 8; t++)
        for (int k = 0; k < topk; k++)
            if (ti[t * topk + k] == neuron) cnt++;
    sc[tid] = cnt; __syncthreads();
    for (int off = 1; off < 256; off <<= 1) {
        int v = (tid >= off) ? sc[tid - off] : 0;
        __syncthreads();
        sc[tid] += v;
        __syncthreads();
    }
    int pos = sc[tid] - cnt;
    int*   bt  = btok + ((size_t)r * NNEUR + neuron) * CAP;
    float* bwp = bw   + ((size_t)r * NNEUR + neuron) * CAP;
    for (int t = t0; t < t0 + 8; t++)
        for (int k = 0; k < topk; k++)
            if (ti[t * topk + k] == neuron) { bt[pos] = t; bwp[pos] = twr[t * topk + k]; pos++; }
    if (tid == 255) bcnt[(size_t)r * NNEUR + neuron] = sc[255];
}

// ---------------- causal attention; K/V tiles staged via async global->LDS ----------------
__global__ void attn_kernel(const float* __restrict__ Q, const float* __restrict__ K,
                            const float* __restrict__ V, float* __restrict__ AO,
                            __bf16* __restrict__ AObf) {
    __shared__ __attribute__((aligned(16))) float sK[64][32];
    __shared__ __attribute__((aligned(16))) float sV[64][32];
    int tid = threadIdx.x;                    // 64
    int bh = blockIdx.y, b = bh >> 3, h = bh & 7;
    int q = blockIdx.x * 64 + tid;
    int t = b * SEQ + q;
    float qr[32], acc[32];
    const float* Qp = Q + (size_t)t * RANK + h * DHEAD;
#pragma unroll
    for (int d = 0; d < 32; d++) { qr[d] = Qp[d]; acc[d] = 0.f; }
    float m = -1e30f, l = 0.f;
    const float isq = 0.17677669529663687f;   // 1/sqrt(32)
    for (int kt = 0; kt <= blockIdx.x; kt++) {
        int t2 = b * SEQ + kt * 64 + tid;
        const float* Kp = K + (size_t)t2 * RANK + h * DHEAD;
        const float* Vp = V + (size_t)t2 * RANK + h * DHEAD;
        // each thread async-copies its 128B K row and 128B V row straight into LDS
        unsigned lK = (unsigned)(uintptr_t)&sK[tid][0];
        unsigned lV = (unsigned)(uintptr_t)&sV[tid][0];
        unsigned long long gK = (unsigned long long)(uintptr_t)Kp;
        unsigned long long gV = (unsigned long long)(uintptr_t)Vp;
        asm volatile(
            "global_load_async_to_lds_b128 %0, %2, off\n\t"
            "global_load_async_to_lds_b128 %0, %2, off offset:16\n\t"
            "global_load_async_to_lds_b128 %0, %2, off offset:32\n\t"
            "global_load_async_to_lds_b128 %0, %2, off offset:48\n\t"
            "global_load_async_to_lds_b128 %0, %2, off offset:64\n\t"
            "global_load_async_to_lds_b128 %0, %2, off offset:80\n\t"
            "global_load_async_to_lds_b128 %0, %2, off offset:96\n\t"
            "global_load_async_to_lds_b128 %0, %2, off offset:112\n\t"
            "global_load_async_to_lds_b128 %1, %3, off\n\t"
            "global_load_async_to_lds_b128 %1, %3, off offset:16\n\t"
            "global_load_async_to_lds_b128 %1, %3, off offset:32\n\t"
            "global_load_async_to_lds_b128 %1, %3, off offset:48\n\t"
            "global_load_async_to_lds_b128 %1, %3, off offset:64\n\t"
            "global_load_async_to_lds_b128 %1, %3, off offset:80\n\t"
            "global_load_async_to_lds_b128 %1, %3, off offset:96\n\t"
            "global_load_async_to_lds_b128 %1, %3, off offset:112\n\t"
            "s_wait_asynccnt 0"
            :: "v"(lK), "v"(lV), "v"(gK), "v"(gV)
            : "memory");
        __syncthreads();
        int kmax = min(64, q - kt * 64 + 1);
        for (int kk = 0; kk < kmax; kk++) {
            float s = 0.f;
#pragma unroll
            for (int d = 0; d < 32; d++) s += qr[d] * sK[kk][d];
            s *= isq;
            float nm = fmaxf(m, s);
            float scl = __expf(m - nm);
            float p   = __expf(s - nm);
            l = l * scl + p;
#pragma unroll
            for (int d = 0; d < 32; d++) acc[d] = acc[d] * scl + p * sV[kk][d];
            m = nm;
        }
        __syncthreads();
    }
    float inv = 1.f / l;
    float*  op = AO   + (size_t)t * RANK + h * DHEAD;
    __bf16* ob = AObf + (size_t)t * RANK + h * DHEAD;
#pragma unroll
    for (int d = 0; d < 32; d++) { float o = acc[d] * inv; op[d] = o; ob[d] = (__bf16)o; }
}

// ---------------- knowledge top-8 over 8192 + weighted gather + residual ----------------
__global__ void ktopk_kernel(const float* __restrict__ KS, const float* __restrict__ KV,
                             const float* __restrict__ X1, float* __restrict__ Out) {
    __shared__ float rv[256]; __shared__ int ri[256]; __shared__ int rt[256];
    __shared__ float topv[8]; __shared__ int topi[8];
    int t = blockIdx.x, tid = threadIdx.x;
    const float* sc = KS + (size_t)t * NKNOW;
    float lv[8]; int li[8];
#pragma unroll
    for (int j = 0; j < 8; j++) { lv[j] = -1e30f; li[j] = NKNOW; }
    for (int e = 0; e < NKNOW / 256; e++) {
        int n = tid + e * 256;
        float v = sc[n];
        if (v > lv[7] || (v == lv[7] && n < li[7])) {
            int j = 7;
            while (j > 0 && (v > lv[j - 1] || (v == lv[j - 1] && n < li[j - 1]))) {
                lv[j] = lv[j - 1]; li[j] = li[j - 1]; j--;
            }
            lv[j] = v; li[j] = n;
        }
    }
    int ptr = 0;
    for (int r = 0; r < 8; r++) {
        rv[tid] = (ptr < 8) ? lv[ptr] : -1e30f;
        ri[tid] = (ptr < 8) ? li[ptr] : NKNOW;
        rt[tid] = tid;
        __syncthreads();
        for (int off = 128; off > 0; off >>= 1) {
            if (tid < off) {
                float v2 = rv[tid + off]; int i2 = ri[tid + off];
                if (v2 > rv[tid] || (v2 == rv[tid] && i2 < ri[tid])) {
                    rv[tid] = v2; ri[tid] = i2; rt[tid] = rt[tid + off];
                }
            }
            __syncthreads();
        }
        if (tid == 0) { topv[r] = rv[0]; topi[r] = ri[0]; }
        int win = rt[0];
        __syncthreads();
        if (tid == win) ptr++;
        __syncthreads();
    }
    float mx = topv[0], ssum = 0.f, wl[8];
#pragma unroll
    for (int j = 0; j < 8; j++) { wl[j] = __expf(topv[j] - mx); ssum += wl[j]; }
    float inv = 1.f / ssum;
    for (int d = tid; d < DMODEL; d += 256) {
        float a = X1[(size_t)t * DMODEL + d];
#pragma unroll
        for (int j = 0; j < 8; j++) a += wl[j] * inv * KV[(size_t)topi[j] * DMODEL + d];
        Out[(size_t)t * DMODEL + d] = a;
    }
}

// ---------------- host launcher ----------------
extern "C" void kernel_launch(void* const* d_in, const int* in_sizes, int n_in,
                              void* d_out, int out_size, void* d_ws, size_t ws_size,
                              hipStream_t stream) {
    const float* x   = (const float*)d_in[0];
    // d_in[1] = mask (unused; causal handled analytically)
    const float* cN  = (const float*)d_in[2];   // [64][1024][256]
    const float* eN  = (const float*)d_in[3];   // [64][256][1024]
    const float* kK  = (const float*)d_in[4];   // [8192][256]
    const float* kV  = (const float*)d_in[5];   // [8192][1024]
    const float* rq  = (const float*)d_in[6];
    const float* rk  = (const float*)d_in[7];
    const float* rv_ = (const float*)d_in[8];
    const float* ro  = (const float*)d_in[9];
    const float* rm  = (const float*)d_in[10];
    const float* g1  = (const float*)d_in[11];
    const float* b1  = (const float*)d_in[12];
    const float* g2  = (const float*)d_in[13];
    const float* b2  = (const float*)d_in[14];
    float* out = (float*)d_out;

    size_t off = 0;
    auto alloc = [&](size_t bytes) -> char* {
        off = (off + 255) & ~(size_t)255;
        char* p = (char*)d_ws + off;
        off += bytes;
        return p;
    };

    // prepacked bf16 weights
    __bf16* pcN  = (__bf16*)alloc((size_t)NNEUR * DMODEL * RANK * 2);
    __bf16* peN  = (__bf16*)alloc((size_t)NNEUR * RANK * DMODEL * 2);
    __bf16* pkK  = (__bf16*)alloc((size_t)NKNOW * RANK * 2);
    __bf16* prq  = (__bf16*)alloc((size_t)NNEUR * DMODEL * 2);
    __bf16* prk  = (__bf16*)alloc((size_t)NNEUR * DMODEL * 2);
    __bf16* prv  = (__bf16*)alloc((size_t)NNEUR * DMODEL * 2);
    __bf16* pro  = (__bf16*)alloc((size_t)NNEUR * RANK * 2);
    __bf16* prm  = (__bf16*)alloc((size_t)NNEUR * DMODEL * 2);
    // activations / intermediates
    __bf16* h_bf   = (__bf16*)alloc((size_t)T_TOK * DMODEL * 2);
    float*  rs_qkv = (float*) alloc((size_t)3 * T_TOK * NNEUR * 4);
    int*    ti_qkv = (int*)   alloc((size_t)3 * T_TOK * 8 * 4);
    float*  tw_qkv = (float*) alloc((size_t)3 * T_TOK * 8 * 4);
    int*    bt_qkv = (int*)   alloc((size_t)3 * NNEUR * CAP * 4);
    float*  bw_qkv = (float*) alloc((size_t)3 * NNEUR * CAP * 4);
    int*    bc_qkv = (int*)   alloc((size_t)3 * NNEUR * 4);
    float*  Qf     = (float*) alloc((size_t)3 * T_TOK * RANK * 4);   // Q,K,V contiguous
    float*  Kf     = Qf + (size_t)T_TOK * RANK;
    float*  Vf     = Qf + (size_t)2 * T_TOK * RANK;
    float*  AO     = (float*) alloc((size_t)T_TOK * RANK * 4);
    __bf16* AObf   = (__bf16*)alloc((size_t)T_TOK * RANK * 2);
    float*  rs_e   = (float*) alloc((size_t)T_TOK * NNEUR * 4);
    int*    ti_e   = (int*)   alloc((size_t)T_TOK * 4 * 4);
    float*  tw_e   = (float*) alloc((size_t)T_TOK * 4 * 4);
    int*    bt_e   = (int*)   alloc((size_t)NNEUR * CAP * 4);
    float*  bw_e   = (float*) alloc((size_t)NNEUR * CAP * 4);
    int*    bc_e   = (int*)   alloc((size_t)NNEUR * 4);
    float*  x1     = (float*) alloc((size_t)T_TOK * DMODEL * 4);
    __bf16* h2bf   = (__bf16*)alloc((size_t)T_TOK * DMODEL * 2);
    float*  rs_m   = (float*) alloc((size_t)T_TOK * NNEUR * 4);
    int*    ti_m   = (int*)   alloc((size_t)T_TOK * 8 * 4);
    float*  tw_m   = (float*) alloc((size_t)T_TOK * 8 * 4);
    int*    bt_m   = (int*)   alloc((size_t)NNEUR * CAP * 4);
    float*  bw_m   = (float*) alloc((size_t)NNEUR * CAP * 4);
    int*    bc_m   = (int*)   alloc((size_t)NNEUR * 4);
    float*  Qm     = (float*) alloc((size_t)T_TOK * RANK * 4);
    __bf16* Qmbf   = (__bf16*)alloc((size_t)T_TOK * RANK * 2);
    float*  kscr   = (float*) alloc((size_t)T_TOK * NKNOW * 4);

    // ---- prepack all weight operands into WMMA fragment layout (one pass each) ----
    prepack_kernel<<<(NNEUR * DMODEL * RANK) / 256, 256, 0, stream>>>(cN, pcN, DMODEL, RANK, 0);
    prepack_kernel<<<(NNEUR * RANK * DMODEL) / 256, 256, 0, stream>>>(eN, peN, RANK, DMODEL, 0);
    prepack_kernel<<<(NKNOW * RANK)         / 256, 256, 0, stream>>>(kK, pkK, RANK, NKNOW, 1);
    prepack_kernel<<<(NNEUR * DMODEL)       / 256, 256, 0, stream>>>(rq, prq, DMODEL, NNEUR, 1);
    prepack_kernel<<<(NNEUR * DMODEL)       / 256, 256, 0, stream>>>(rk, prk, DMODEL, NNEUR, 1);
    prepack_kernel<<<(NNEUR * DMODEL)       / 256, 256, 0, stream>>>(rv_, prv, DMODEL, NNEUR, 1);
    prepack_kernel<<<(NNEUR * RANK)         / 256, 256, 0, stream>>>(ro, pro, RANK, NNEUR, 1);
    prepack_kernel<<<(NNEUR * DMODEL)       / 256, 256, 0, stream>>>(rm, prm, DMODEL, NNEUR, 1);

    const __bf16* prqkv[3] = { prq, prk, prv };

    // ---- attention sub-block ----
    ln_bf16_kernel<<<T_TOK, 256, 0, stream>>>(x, g1, b1, h_bf);

    for (int r = 0; r < 3; r++)   // router scores via dense wmma gemm on packed B
        wmma_gemm_kernel<false><<<dim3(1, T_TOK / 16, 1), 128, 0, stream>>>(
            h_bf, prqkv[r], rs_qkv + (size_t)r * T_TOK * NNEUR,
            nullptr, nullptr, nullptr, DMODEL, NNEUR, NNEUR, 1.0f);

    topk_softmax_kernel<<<dim3(T_TOK, 3), 32, 0, stream>>>(rs_qkv, 8, ti_qkv, tw_qkv);
    bin_kernel<<<dim3(NNEUR, 3), 256, 0, stream>>>(ti_qkv, tw_qkv, 8, bt_qkv, bw_qkv, bc_qkv);

    {   int n = 3 * T_TOK * RANK;
        zero_f32_kernel<<<(n + 255) / 256, 256, 0, stream>>>(Qf, n); }

    float* qkvOut[3] = { Qf, Kf, Vf };
    for (int r = 0; r < 3; r++)   // gathered compress GEMMs (K=1024, N=256)
        wmma_gemm_kernel<true><<<dim3(RANK / 64, CAP / 16, NNEUR), 128, 0, stream>>>(
            h_bf, pcN, qkvOut[r],
            bt_qkv + (size_t)r * NNEUR * CAP, bw_qkv + (size_t)r * NNEUR * CAP,
            bc_qkv + (size_t)r * NNEUR, DMODEL, RANK, RANK, 1.0f);

    attn_kernel<<<dim3(SEQ / 64, 4 * NHEAD), 64, 0, stream>>>(Qf, Kf, Vf, AO, AObf);

    // ---- expand ----
    wmma_gemm_kernel<false><<<dim3(1, T_TOK / 16, 1), 128, 0, stream>>>(
        AObf, pro, rs_e, nullptr, nullptr, nullptr, RANK, NNEUR, NNEUR, 1.0f);
    topk_softmax_kernel<<<dim3(T_TOK, 1), 32, 0, stream>>>(rs_e, 4, ti_e, tw_e);
    bin_kernel<<<dim3(NNEUR, 1), 256, 0, stream>>>(ti_e, tw_e, 4, bt_e, bw_e, bc_e);

    {   int n = T_TOK * DMODEL;
        copy_f32_kernel<<<(n + 255) / 256, 256, 0, stream>>>(x1, x, n); }

    wmma_gemm_kernel<true><<<dim3(DMODEL / 64, CAP / 16, NNEUR), 128, 0, stream>>>(
        AObf, peN, x1, bt_e, bw_e, bc_e, RANK, DMODEL, DMODEL, 1.0f);

    // ---- memory sub-block ----
    ln_bf16_kernel<<<T_TOK, 256, 0, stream>>>(x1, g2, b2, h2bf);

    wmma_gemm_kernel<false><<<dim3(1, T_TOK / 16, 1), 128, 0, stream>>>(
        h2bf, prm, rs_m, nullptr, nullptr, nullptr, DMODEL, NNEUR, NNEUR, 1.0f);
    topk_softmax_kernel<<<dim3(T_TOK, 1), 32, 0, stream>>>(rs_m, 8, ti_m, tw_m);
    bin_kernel<<<dim3(NNEUR, 1), 256, 0, stream>>>(ti_m, tw_m, 8, bt_m, bw_m, bc_m);

    {   int n = T_TOK * RANK;
        zero_f32_kernel<<<(n + 255) / 256, 256, 0, stream>>>(Qm, n); }

    wmma_gemm_kernel<true><<<dim3(RANK / 64, CAP / 16, NNEUR), 128, 0, stream>>>(
        h2bf, pcN, Qm, bt_m, bw_m, bc_m, DMODEL, RANK, RANK, 1.0f);

    {   int n = T_TOK * RANK;
        f32_to_bf16_kernel<<<(n + 255) / 256, 256, 0, stream>>>(Qmbf, Qm, n); }

    // knowledge scores: [2048x256] @ [256x8192] packed, scaled 1/sqrt(256)
    wmma_gemm_kernel<false><<<dim3(NKNOW / 64, T_TOK / 16, 1), 128, 0, stream>>>(
        Qmbf, pkK, kscr, nullptr, nullptr, nullptr, RANK, NKNOW, NKNOW, 0.0625f);

    ktopk_kernel<<<T_TOK, 256, 0, stream>>>(kscr, kV, x1, out);
}
